// BigramLanguageModel_30193620091027
// MI455X (gfx1250) — matrix operand order repaired and checked by reading, not compiled, not run on hardware
//
#include <hip/hip_runtime.h>
#include <hip/hip_bf16.h>

// ---------------------------------------------------------------------------
// Types / helpers for CDNA5 WMMA (gfx1250, wave32)
// ---------------------------------------------------------------------------
typedef unsigned short u16;
typedef __attribute__((ext_vector_type(8)))  u16    v8u;
typedef __attribute__((ext_vector_type(16))) u16    v16u;
typedef __attribute__((ext_vector_type(16))) __bf16 v16bf;
typedef __attribute__((ext_vector_type(8)))  float  v8f;

#define BB 2
#define TT 2048
#define EE 1024
#define HH 16
#define DD 64
#define VV 32000
#define MM (BB*TT)          // 4096 rows

__device__ __forceinline__ u16 f2bf(float f) {
  union { float f; unsigned u; } x; x.f = f;
  unsigned r = x.u + 0x7FFFu + ((x.u >> 16) & 1u);   // round-to-nearest-even
  return (u16)(r >> 16);
}

__device__ __forceinline__ v8u ld8(const u16* p)  { return *(const v8u*)p; }
__device__ __forceinline__ v16u ld16(const u16* p){ return *(const v16u*)p; }

__device__ __forceinline__ v16u cat8(v8u lo, v8u hi) {
  return __builtin_shufflevector(lo, hi, 0,1,2,3,4,5,6,7,8,9,10,11,12,13,14,15);
}

__device__ __forceinline__ v8f wmma_bf16(v16u a, v16u b, v8f c) {
  return __builtin_amdgcn_wmma_f32_16x16x32_bf16(
      /*neg_a=*/false, __builtin_bit_cast(v16bf, a),
      /*neg_b=*/false, __builtin_bit_cast(v16bf, b),
      /*c_mod=*/(short)0, c, /*reuse_a=*/false, /*reuse_b=*/false);
}

__device__ __forceinline__ float halfwave_max(float v) {
  #pragma unroll
  for (int o = 8; o >= 1; o >>= 1) v = fmaxf(v, __shfl_xor(v, o, 32));
  return v;
}
__device__ __forceinline__ float halfwave_sum(float v) {
  #pragma unroll
  for (int o = 8; o >= 1; o >>= 1) v += __shfl_xor(v, o, 32);
  return v;
}

// Async global->LDS copy, 16B per lane, GVS addressing (SGPR base + VGPR offset).
// Tracked by ASYNCcnt; drained with s_wait_asynccnt.
__device__ __forceinline__ void async_ld_b128(unsigned lds_byte_addr,
                                              unsigned glb_byte_off,
                                              const u16* base) {
  asm volatile("global_load_async_to_lds_b128 %0, %1, %2 offset:0"
               :: "v"(lds_byte_addr), "v"(glb_byte_off), "s"(base) : "memory");
}
__device__ __forceinline__ void wait_async0() {
  asm volatile("s_wait_asynccnt 0" ::: "memory");
}
__device__ __forceinline__ unsigned lds_addr_of(const void* p) {
  // LDS aperture: hardware uses addr[31:0] as the LDS byte address
  return (unsigned)(size_t)p;
}

// ---------------------------------------------------------------------------
// 1) fp32 -> bf16 conversion
// ---------------------------------------------------------------------------
__global__ void cvt_bf16_kernel(const float* __restrict__ in, u16* __restrict__ out, int n) {
  int i = blockIdx.x * 256 + threadIdx.x;
  if (i < n) out[i] = f2bf(in[i]);
}

// ---------------------------------------------------------------------------
// 2) embedding: x[b,t,:] = tok_table[tokens[b,t]] + pos_emb[t]  (-> bf16)
// ---------------------------------------------------------------------------
__global__ void embed_kernel(const int* __restrict__ tokens,
                             const float* __restrict__ tok_table,
                             const float* __restrict__ pos_emb,
                             u16* __restrict__ xb) {
  int i = blockIdx.x * 256 + threadIdx.x;          // over 4096*1024
  int m = i >> 10, e = i & (EE - 1);
  int t = m & (TT - 1);
  int tok = tokens[m];
  float v = tok_table[(size_t)tok * EE + e] + pos_emb[(size_t)t * EE + e];
  xb[i] = f2bf(v);
}

// ---------------------------------------------------------------------------
// 3) QKV projection GEMM: out[b,h,t,d] = sum_e x[b,t,e]*W[h,d,e] + bias[h,d]
//    One 16x16 output tile per wave, K=1024 in steps of 32 (bf16 WMMA).
// ---------------------------------------------------------------------------
__global__ __launch_bounds__(256) void gemm_qkv_kernel(
    const u16* __restrict__ X,     // [4096,1024] bf16
    const u16* __restrict__ W,     // [1024,1024] bf16 (row n = h*64+d, col e)
    const float* __restrict__ bias,// [1024]
    u16* __restrict__ out)         // [B,H,T,D] bf16
{
  const int lane = threadIdx.x & 31, wv = threadIdx.x >> 5;
  const int li = lane & 15, hi = lane >> 4;
  const int tile = blockIdx.x * 8 + wv;            // 16384 tiles
  const int mt = tile >> 6, nt = tile & 63;        // 256 x 64 tiles
  const int m0 = mt * 16, n0 = nt * 16;

  const u16* arow = X + (size_t)(m0 + li) * EE;
  const u16* wrow = W + (size_t)(n0 + li) * EE;

  v8f acc = {0,0,0,0,0,0,0,0};
  #pragma unroll 4
  for (int kk = 0; kk < EE; kk += 32) {
    v16u a = ld16(arow + kk + hi * 16);
    v16u b = ld16(wrow + kk + hi * 16);
    acc = wmma_bf16(a, b, acc);
  }

  const int n = n0 + li;
  const float bn = bias[n];
  const int h = n >> 6, d = n & 63;
  #pragma unroll
  for (int r = 0; r < 8; ++r) {
    int m = m0 + r + 8 * hi;
    int b_ = m >> 11, t = m & (TT - 1);
    out[((((size_t)b_ * HH + h) * TT + t) * DD) + d] = f2bf(acc[r] + bn);
  }
}

// ---------------------------------------------------------------------------
// 4) Flash attention (quirky reference semantics):
//    mask: excluded if (s > t) OR raw score == 0.0; softmax; then *1/sqrt(D).
// ---------------------------------------------------------------------------
__global__ __launch_bounds__(128) void attn_kernel(
    const u16* __restrict__ Q,   // [B,H,T,D] bf16
    const u16* __restrict__ K,
    const u16* __restrict__ V,
    u16* __restrict__ O)         // [B,T,E] bf16 (head-concat)
{
  __shared__ u16 sP[4][16 * 16];
  __shared__ u16 sVT[4][DD * 16];

  const int lane = threadIdx.x & 31, wv = threadIdx.x >> 5;
  const int li = lane & 15, hi = lane >> 4;
  const int w = blockIdx.x * 4 + wv;               // 4096 waves
  const int qblk = w & 127;                        // T/16 = 128
  const int h = (w >> 7) & 15;
  const int b = w >> 11;
  const int t0 = qblk * 16;
  const float NEG = -__builtin_inff();

  const u16* qp = Q + (((size_t)b * HH + h) * TT) * DD;
  const u16* kp = K + (((size_t)b * HH + h) * TT) * DD;
  const u16* vp = V + (((size_t)b * HH + h) * TT) * DD;

  const u16* qrow = qp + (size_t)(t0 + li) * DD;
  v16u a0 = ld16(qrow + hi * 16);
  v16u a1 = ld16(qrow + 32 + hi * 16);

  float run_m[8], run_l[8];
  v8f acc[4];
  #pragma unroll
  for (int r = 0; r < 8; ++r) { run_m[r] = NEG; run_l[r] = 0.f; }
  v8f zacc = {0,0,0,0,0,0,0,0};
  #pragma unroll
  for (int c = 0; c < 4; ++c) acc[c] = zacc;
  v8u z8 = {0,0,0,0,0,0,0,0};

  for (int ks = 0; ks <= qblk; ++ks) {
    const int s0 = ks * 16;

    const u16* krow = kp + (size_t)(s0 + li) * DD;
    v16u b0 = ld16(krow + hi * 16);
    v16u b1 = ld16(krow + 32 + hi * 16);
    v8f s = wmma_bf16(a1, b1, wmma_bf16(a0, b0, zacc));

    const u16* vrow = vp + (size_t)(s0 + li) * DD + hi * 32;
    v8u v0 = ld8(vrow), v1 = ld8(vrow + 8), v2 = ld8(vrow + 16), v3 = ld8(vrow + 24);
    #pragma unroll
    for (int j = 0; j < 8; ++j) {
      sVT[wv][(hi * 32 + j) * 16 + li]      = v0[j];
      sVT[wv][(hi * 32 + 8 + j) * 16 + li]  = v1[j];
      sVT[wv][(hi * 32 + 16 + j) * 16 + li] = v2[j];
      sVT[wv][(hi * 32 + 24 + j) * 16 + li] = v3[j];
    }

    #pragma unroll
    for (int r = 0; r < 8; ++r) {
      const int trow = t0 + r + 8 * hi;
      const int scol = s0 + li;
      float sc = s[r];
      bool ex = (scol > trow) || (sc == 0.0f);     // faithful quirk
      float val = ex ? NEG : sc;
      float tm = halfwave_max(val);
      float nm = fmaxf(run_m[r], tm);
      float corr = (nm == run_m[r]) ? 1.0f : __expf(run_m[r] - nm);
      float p = ex ? 0.0f : __expf(sc - nm);
      run_l[r] = run_l[r] * corr + halfwave_sum(p);
      run_m[r] = nm;
      #pragma unroll
      for (int c = 0; c < 4; ++c) acc[c][r] *= corr;
      sP[wv][(r + 8 * hi) * 16 + li] = f2bf(p);
    }

    v16u pa = (hi == 0) ? ld16(&sP[wv][li * 16]) : cat8(z8, z8);

    #pragma unroll
    for (int c = 0; c < 4; ++c) {
      v16u bv;
      if (hi == 0) bv = ld16(&sVT[wv][(c * 16 + li) * 16]);
      else         bv = cat8(z8, z8);
      acc[c] = wmma_bf16(pa, bv, acc[c]);
    }
  }

  #pragma unroll
  for (int r = 0; r < 8; ++r) {
    float inv = 1.0f / (run_l[r] * 8.0f);          // sqrt(64) = 8
    int t = t0 + r + 8 * hi;
    #pragma unroll
    for (int c = 0; c < 4; ++c) {
      O[(((size_t)b * TT + t) * EE) + h * DD + c * 16 + li] = f2bf(acc[c][r] * inv);
    }
  }
}

// ---------------------------------------------------------------------------
// 5) logits GEMM (dominant: 268 GFLOP): block-tiled 64x128, K-steps of 32,
//    operands staged global->LDS via async b128 loads, double buffered.
//    8 waves/block, each wave: 1 Mtile x 4 Ntiles -> 4 WMMAs per K-step.
// ---------------------------------------------------------------------------
__global__ __launch_bounds__(256) void gemm_logits_kernel(
    const u16* __restrict__ O,     // [4096,1024] bf16
    const u16* __restrict__ W,     // [32000,1024] bf16
    const float* __restrict__ bo,  // [32000]
    float* __restrict__ out)       // [4096,32000] fp32
{
  __shared__ u16 sA[2][64 * 32];    //  8 KB  A tile: 64 rows x 32 k
  __shared__ u16 sB[2][128 * 32];   // 16 KB  B tile: 128 rows x 32 k

  const int tid  = threadIdx.x;
  const int lane = tid & 31, wv = tid >> 5;
  const int li = lane & 15, hi = lane >> 4;
  const int m0 = blockIdx.y * 64;                  // 64 mrows/block
  const int n0 = blockIdx.x * 128;                 // 128 ncols/block

  __builtin_amdgcn_s_wait_tensorcnt(0);

  const unsigned sA0 = lds_addr_of(&sA[0][0]);
  const unsigned sA1 = lds_addr_of(&sA[1][0]);
  const unsigned sB0 = lds_addr_of(&sB[0][0]);
  const unsigned sB1 = lds_addr_of(&sB[1][0]);

  // per-thread async-copy assignments (16B = 8 bf16 per chunk)
  const int ar = tid >> 2, aseg = tid & 3;             // A: 64 rows x 4 segs
  const unsigned aLds = (unsigned)(ar * 32 + aseg * 8) * 2;
  const unsigned aGlb0 = (unsigned)((m0 + ar) * EE + aseg * 8) * 2;
  const int br0 = tid >> 1, bseg0 = (tid & 1) * 2;     // B: 128 rows x 4 segs, 2/thread
  const unsigned bLds0 = (unsigned)(br0 * 32 + bseg0 * 8) * 2;
  const unsigned bGlb0 = (unsigned)((n0 + br0) * EE + bseg0 * 8) * 2;

  // prologue: fill buffer 0 for kk=0
  async_ld_b128(sA0 + aLds, aGlb0, O);
  async_ld_b128(sB0 + bLds0,      bGlb0,      W);
  async_ld_b128(sB0 + bLds0 + 16, bGlb0 + 16, W);
  wait_async0();
  __syncthreads();

  const int mi = wv >> 1;                          // 0..3
  const int nb = (wv & 1) * 4;                     // 0 or 4 (x16 cols)
  v8f acc[4];
  v8f zacc = {0,0,0,0,0,0,0,0};
  #pragma unroll
  for (int t = 0; t < 4; ++t) acc[t] = zacc;

  int buf = 0;
  for (int kk = 0; kk < EE; kk += 32) {
    // issue async fills of the other buffer for the next K-step
    if (kk + 32 < EE) {
      const unsigned nA = buf ? sA0 : sA1;
      const unsigned nB = buf ? sB0 : sB1;
      const unsigned koff = (unsigned)(kk + 32) * 2;
      async_ld_b128(nA + aLds, aGlb0 + koff, O);
      async_ld_b128(nB + bLds0,      bGlb0 + koff,      W);
      async_ld_b128(nB + bLds0 + 16, bGlb0 + koff + 16, W);
    }

    // compute from current buffer
    const u16* sar = &sA[buf][(mi * 16 + li) * 32];
    v16u a = ld16(sar + hi * 16);
    #pragma unroll
    for (int t = 0; t < 4; ++t) {
      v16u b = ld16(&sB[buf][((nb + t) * 16 + li) * 32 + hi * 16]);
      acc[t] = wmma_bf16(a, b, acc[t]);
    }

    wait_async0();        // this wave's fills of the other buffer done
    __syncthreads();      // everyone done reading buf + filling buf^1
    buf ^= 1;
  }

  // store 4 tiles of 16x16 fp32 (+ bias)
  #pragma unroll
  for (int t = 0; t < 4; ++t) {
    const int n = n0 + (nb + t) * 16 + li;
    const float bn = bo[n];
    #pragma unroll
    for (int r = 0; r < 8; ++r) {
      int m = m0 + mi * 16 + r + 8 * hi;
      out[(size_t)m * VV + n] = acc[t][r] + bn;
    }
  }
}

// ---------------------------------------------------------------------------
// 6) loss: per-row log-softmax + cross entropy, mean over 4096 rows
// ---------------------------------------------------------------------------
__global__ void zero_loss_kernel(float* loss) { *loss = 0.0f; }

__global__ __launch_bounds__(256) void loss_kernel(
    const float* __restrict__ logits, const int* __restrict__ targets,
    float* __restrict__ loss)
{
  const int row = blockIdx.x, tid = threadIdx.x;
  const float* lr = logits + (size_t)row * VV;
  __shared__ float red[256];

  float mx = -__builtin_inff();
  for (int i = tid; i < VV; i += 256) mx = fmaxf(mx, lr[i]);
  red[tid] = mx; __syncthreads();
  for (int s = 128; s > 0; s >>= 1) { if (tid < s) red[tid] = fmaxf(red[tid], red[tid + s]); __syncthreads(); }
  mx = red[0]; __syncthreads();

  float sum = 0.f;
  for (int i = tid; i < VV; i += 256) sum += __expf(lr[i] - mx);
  red[tid] = sum; __syncthreads();
  for (int s = 128; s > 0; s >>= 1) { if (tid < s) red[tid] += red[tid + s]; __syncthreads(); }

  if (tid == 0) {
    int t = targets[row];
    float lp = lr[t] - mx - __logf(red[0]);
    atomicAdd(loss, -lp * (1.0f / (float)MM));
  }
}

// ---------------------------------------------------------------------------
// Host launcher
// ---------------------------------------------------------------------------
extern "C" void kernel_launch(void* const* d_in, const int* in_sizes, int n_in,
                              void* d_out, int out_size, void* d_ws, size_t ws_size,
                              hipStream_t stream) {
  const int*   tokens    = (const int*)  d_in[0];
  const int*   targets   = (const int*)  d_in[1];
  const float* tok_table = (const float*)d_in[2];
  const float* pos_emb   = (const float*)d_in[3];
  const float* Wq        = (const float*)d_in[4];
  const float* bq        = (const float*)d_in[5];
  const float* Wk        = (const float*)d_in[6];
  const float* bk        = (const float*)d_in[7];
  const float* Wv        = (const float*)d_in[8];
  const float* bv        = (const float*)d_in[9];
  const float* Wo        = (const float*)d_in[10];
  const float* bo        = (const float*)d_in[11];

  char* ws = (char*)d_ws;
  u16* xb   = (u16*)(ws + 0);                       //  8 MB  [4096,1024]
  u16* qb   = (u16*)(ws + (8u  << 20));             //  8 MB  [B,H,T,D]
  u16* kb   = (u16*)(ws + (16u << 20));             //  8 MB
  u16* vb   = (u16*)(ws + (24u << 20));             //  8 MB
  u16* ob   = (u16*)(ws + (32u << 20));             //  8 MB  [4096,1024]
  u16* wqb  = (u16*)(ws + (40u << 20));             //  2 MB
  u16* wkb  = (u16*)(ws + (42u << 20));             //  2 MB
  u16* wvb  = (u16*)(ws + (44u << 20));             //  2 MB
  u16* wob  = (u16*)(ws + (46u << 20));             // 62.5 MB [32000,1024]

  float* logits = (float*)d_out;                    // [4096,32000]
  float* loss   = logits + (size_t)MM * VV;         // final element

  cvt_bf16_kernel<<<(EE*EE)/256, 256, 0, stream>>>(Wq, wqb, EE*EE);
  cvt_bf16_kernel<<<(EE*EE)/256, 256, 0, stream>>>(Wk, wkb, EE*EE);
  cvt_bf16_kernel<<<(EE*EE)/256, 256, 0, stream>>>(Wv, wvb, EE*EE);
  cvt_bf16_kernel<<<(VV*EE)/256, 256, 0, stream>>>(Wo, wob, VV*EE);

  embed_kernel<<<(MM*EE)/256, 256, 0, stream>>>(tokens, tok_table, pos_emb, xb);

  gemm_qkv_kernel<<<2048, 256, 0, stream>>>(xb, wqb, bq, qb);
  gemm_qkv_kernel<<<2048, 256, 0, stream>>>(xb, wkb, bk, kb);
  gemm_qkv_kernel<<<2048, 256, 0, stream>>>(xb, wvb, bv, vb);

  attn_kernel<<<1024, 128, 0, stream>>>(qb, kb, vb, ob);

  // logits GEMM: grid = (32000/128) x (4096/64)
  dim3 lgrid(VV / 128, MM / 64);
  gemm_logits_kernel<<<lgrid, 256, 0, stream>>>(ob, wob, bo, logits);

  zero_loss_kernel<<<1, 1, 0, stream>>>(loss);
  loss_kernel<<<MM, 256, 0, stream>>>(logits, targets, loss);
}
